// E3DLSTM_74337293959339
// MI455X (gfx1250) — matrix operand order repaired and checked by reading, not compiled, not run on hardware
//
#include <hip/hip_runtime.h>
#include <hip/hip_bf16.h>
#include <math.h>

// ---------------- model constants ----------------
#define B_   4
#define C0   16
#define HIDN 64
#define T_   4
#define H_   32
#define W_   32
#define SEQ_ 5
#define NL_  2
#define TAU_ 4
#define S_   (T_*H_*W_)      // 4096
#define CS_  (HIDN*S_)       // 262144
#define BCS_ (B_*CS_)        // 1048576
#define TO_  2
#define HO_  30
#define WO_  30
#define PPI_ (TO_*HO_*WO_)   // 1800 positions per image (native conv res)
#define NPOS_ (B_*PPI_)      // 7200

typedef __attribute__((ext_vector_type(16))) __bf16 v16bf;
typedef __attribute__((ext_vector_type(8)))  float  v8f;
typedef __attribute__((ext_vector_type(4)))  unsigned u32x4;

union Frag { v16bf v; u32x4 q[2]; };

// ---------------- WMMA implicit-GEMM conv3d (VALID), fully constant geometry ----
// A: bf16 weights [64][KPAD], k = tap*CIN + ci (tap-major, ci-minor, zero padded)
// X: bf16 activations NDHWC [B][T_][H_][W_][CIN]
// out: fp32 [B][64][To*Ho*Wo]; accumulate==1 -> out += ; else out = +bias
template<int CIN, int CSH, int NTAPS, int KDIM, int KPAD, int TOo, int HOo, int WOo>
__global__ __launch_bounds__(256) void conv_wmma_kernel(
    const __bf16* __restrict__ A, const __bf16* __restrict__ X,
    const float* __restrict__ bias, float* __restrict__ out, int accumulate)
{
  constexpr int HWin = H_ * W_;          // 1024
  constexpr int ppi  = TOo * HOo * WOo;
  constexpr int npos = B_ * ppi;

  const int lane  = threadIdx.x & 31;
  const int wv    = threadIdx.x >> 5;
  const int m0    = (wv & 3) * 16;       // M tile (Cout)
  const int nh    = wv >> 2;             // which N half of the 128-wide block tile
  const int khalf = lane >> 4;
  const int l16   = lane & 15;

  // A-matrix per-lane row (contiguous K order: lanes0-15 K0-15, lanes16-31 K16-31)
  const __bf16* arow = A + (size_t)(m0 + l16) * KPAD + khalf * 16;

  const int nbase = blockIdx.x * 128 + nh * 64;
  int spat[4], bidx[4], prem[4];
#pragma unroll
  for (int j = 0; j < 4; ++j) {
    int n  = nbase + j * 16 + l16;
    int nc = n < npos ? n : npos - 1;
    int b  = nc / ppi;                   // constant divisors -> mul/shift
    int pr = nc - b * ppi;
    int to = pr / (HOo * WOo);
    int r2 = pr - to * (HOo * WOo);
    int ho = r2 / WOo;
    int wo = r2 - ho * WOo;
    bidx[j] = b; prem[j] = pr;
    spat[j] = ((b * T_ + to) * H_ + ho) * W_ + wo;
  }

  const v8f vz = {0.f,0.f,0.f,0.f,0.f,0.f,0.f,0.f};
  v8f acc[4];
#pragma unroll
  for (int j = 0; j < 4; ++j) acc[j] = vz;

  for (int kb = 0; kb < KPAD; kb += 32) {
    Frag af;
    af.q[0] = *(const u32x4*)(arow + kb);
    af.q[1] = *(const u32x4*)(arow + kb + 8);
    if (kb + 32 < KPAD)                  // stream next weight chunk into near cache
      __builtin_prefetch((const void*)(arow + kb + 32), 0, 3);

    const int k0  = kb + khalf * 16;     // lane-group's K base (multiple of 16)
    const int tap = k0 >> CSH;           // run of 16 K stays within one tap (CIN>=16)
    const int ci0 = k0 - (tap << CSH);
    bool valid = true;
    if constexpr (KDIM != KPAD) valid = (tap < NTAPS);
    const int kt = tap / 9;
    const int r3 = tap - kt * 9;
    const int kh = r3 / 3;
    const int kw = r3 - kh * 3;
    const int toff = kt * HWin + kh * W_ + kw;

#pragma unroll
    for (int j = 0; j < 4; ++j) {
      Frag bf;
      if (valid) {
        const __bf16* bp = X + (size_t)(spat[j] + toff) * CIN + ci0;
        bf.q[0] = *(const u32x4*)(bp);
        bf.q[1] = *(const u32x4*)(bp + 8);
      } else {
        const u32x4 z4 = {0u,0u,0u,0u};
        bf.q[0] = z4; bf.q[1] = z4;
      }
      acc[j] = __builtin_amdgcn_wmma_f32_16x16x32_bf16(
          false, af.v, false, bf.v, (short)0, acc[j], false, false);
    }
  }

#pragma unroll
  for (int j = 0; j < 4; ++j) {
    int n = nbase + j * 16 + l16;
    if (n >= npos) continue;
    const size_t obase = ((size_t)bidx[j] * 64) * ppi + prem[j];
#pragma unroll
    for (int v = 0; v < 8; ++v) {
      int cout = m0 + v + khalf * 8;     // C/D layout: lanes16-31 hold M=v+8
      size_t oaddr = obase + (size_t)cout * ppi;
      float val = acc[j][v];
      if (accumulate) out[oaddr] += val;
      else            out[oaddr] = val + (bias ? bias[cout] : 0.f);
    }
  }
}

// three instantiations used by the model
#define CONV_X16   conv_wmma_kernel<16, 4, 27, 432, 448, TO_, HO_, WO_>
#define CONV_H64   conv_wmma_kernel<64, 6, 27, 1728, 1728, TO_, HO_, WO_>
#define CONV_1X1   conv_wmma_kernel<128, 7, 1, 128, 128, T_, H_, W_>

// ---------------- reductions / elementwise ----------------
__device__ __forceinline__ void blockReduce2(float& s, float& s2) {
  __shared__ float sh1[256];
  __shared__ float sh2[256];
  int tid = threadIdx.x;
  sh1[tid] = s; sh2[tid] = s2;
  __syncthreads();
  for (int off = 128; off > 0; off >>= 1) {
    if (tid < off) { sh1[tid] += sh1[tid + off]; sh2[tid] += sh2[tid + off]; }
    __syncthreads();
  }
  s = sh1[0]; s2 = sh2[0];
  __syncthreads();
}

// nearest-resize (2,30,30)->(4,32,32), LayerNorm over THW per (b,c), then act
__global__ __launch_bounds__(256) void gate_ln_act_kernel(
    const float* __restrict__ pre, float* __restrict__ gate, int act)
{
  const int bc = blockIdx.x;                       // b*64+c
  const float* src = pre + (size_t)bc * PPI_;
  float* dst = gate + (size_t)bc * S_;
  float vals[16];
  float s = 0.f, s2 = 0.f;
#pragma unroll
  for (int i = 0; i < 16; ++i) {
    int sidx = threadIdx.x + i * 256;
    int t  = sidx >> 10;
    int hw = sidx & 1023;
    int h  = hw >> 5;
    int w  = hw & 31;
    int st = (t * TO_) >> 2;                       // floor(t*To/T)
    int sh = (h * HO_) >> 5;
    int sw = (w * WO_) >> 5;
    float v = src[(st * HO_ + sh) * WO_ + sw];
    vals[i] = v; s += v; s2 += v * v;
  }
  blockReduce2(s, s2);
  float mean = s * (1.f / S_);
  float var  = s2 * (1.f / S_) - mean * mean;
  float rinv = rsqrtf(var + 1e-5f);
#pragma unroll
  for (int i = 0; i < 16; ++i) {
    float z = (vals[i] - mean) * rinv;
    float g = act ? tanhf(z) : 1.f / (1.f + __expf(-z));
    dst[threadIdx.x + i * 256] = g;
  }
}

__global__ __launch_bounds__(256) void attn_scores_kernel(
    const float* __restrict__ r, const float* __restrict__ histL,
    float* __restrict__ scores)
{
  const int b  = blockIdx.x >> 2;
  const int sl = blockIdx.x & 3;
  const float* rp = r + (size_t)b * CS_;
  const float* hp = histL + (size_t)sl * BCS_ + (size_t)b * CS_;
  float s = 0.f, d = 0.f;
  for (int i = threadIdx.x; i < CS_; i += 256) s += rp[i] * hp[i];
  blockReduce2(s, d);
  if (threadIdx.x == 0) scores[blockIdx.x] = s * (1.0f / 64.0f);  // 1/sqrt(T*H*W)
}

__global__ void softmax_tau_kernel(const float* __restrict__ sc, float* __restrict__ attn)
{
  int b = blockIdx.x;
  if (threadIdx.x == 0) {
    float m = sc[b*4];
    for (int l = 1; l < 4; ++l) m = fmaxf(m, sc[b*4+l]);
    float e[4], sum = 0.f;
    for (int l = 0; l < 4; ++l) { e[l] = __expf(sc[b*4+l] - m); sum += e[l]; }
    float rs = 1.f / sum;
    for (int l = 0; l < 4; ++l) attn[b*4+l] = e[l] * rs;
  }
}

__global__ __launch_bounds__(256) void recall_kernel(
    const float* __restrict__ histL, const float* __restrict__ attn,
    float* __restrict__ rec)
{
  size_t i = (size_t)blockIdx.x * 256 + threadIdx.x;
  if (i >= (size_t)BCS_) return;
  int b = (int)(i / CS_);
  size_t off = i - (size_t)b * CS_;
  const float* hb = histL + (size_t)b * CS_ + off;
  rec[i] = attn[b*4+0] * hb[0]
         + attn[b*4+1] * hb[(size_t)BCS_]
         + attn[b*4+2] * hb[2ull * BCS_]
         + attn[b*4+3] * hb[3ull * BCS_];
}

// LN over (c,t,h,w) per batch of (clast+recall), affine g,b [CS_]
__global__ __launch_bounds__(256) void ln4_kernel(
    const float* __restrict__ clast, const float* __restrict__ rec,
    const float* __restrict__ gam, const float* __restrict__ bet,
    float* __restrict__ outp)
{
  const int b = blockIdx.x;
  const float* z1 = clast + (size_t)b * CS_;
  const float* z2 = rec   + (size_t)b * CS_;
  float s = 0.f, s2 = 0.f;
  for (int i = threadIdx.x; i < CS_; i += 256) {
    float z = z1[i] + z2[i];
    s += z; s2 += z * z;
  }
  blockReduce2(s, s2);
  float mean = s * (1.f / CS_);
  float var  = s2 * (1.f / CS_) - mean * mean;
  float rinv = rsqrtf(var + 1e-5f);
  float* op = outp + (size_t)b * CS_;
  for (int i = threadIdx.x; i < CS_; i += 256) {
    float z = z1[i] + z2[i];
    op[i] = (z - mean) * rinv * gam[i] + bet[i];
  }
}

__global__ __launch_bounds__(256) void fma3_kernel(
    const float* __restrict__ a, const float* __restrict__ b,
    const float* __restrict__ c, float* __restrict__ o, int n)
{ int i = blockIdx.x * 256 + threadIdx.x; if (i < n) o[i] = a[i]*b[i] + c[i]; }

__global__ __launch_bounds__(256) void m_update_kernel(
    const float* __restrict__ ip, const float* __restrict__ gp,
    const float* __restrict__ fp, const float* __restrict__ mp,
    float* __restrict__ m, int n)
{ int i = blockIdx.x * 256 + threadIdx.x; if (i < n) m[i] = ip[i]*gp[i] + fp[i]*mp[i]; }

__global__ __launch_bounds__(256) void h_update_kernel(
    const float* __restrict__ o, const float* __restrict__ cv,
    float* __restrict__ h, int n)
{ int i = blockIdx.x * 256 + threadIdx.x; if (i < n) h[i] = o[i] * tanhf(cv[i]); }

__global__ __launch_bounds__(256) void fill0_kernel(float* p, size_t n)
{ size_t i = (size_t)blockIdx.x * 256 + threadIdx.x; if (i < n) p[i] = 0.f; }

// fp32 NCHW[b][C][s] -> bf16 NDHWC[b][s][Cstride] at channel offset coff
__global__ __launch_bounds__(256) void cvt_ndhwc_kernel(
    const float* __restrict__ in, __bf16* __restrict__ outp,
    int C, int Cstride, int coff)
{
  size_t i = (size_t)blockIdx.x * 256 + threadIdx.x;
  size_t tot = (size_t)B_ * C * S_;
  if (i >= tot) return;
  int cs = C * S_;
  int b = (int)(i / cs);
  int r = (int)(i - (size_t)b * cs);
  int c = r / S_;
  int s = r - c * S_;
  outp[((size_t)b * S_ + s) * Cstride + coff + c] = (__bf16)in[i];
}

// fp32 [64][Cin][ntaps] -> bf16 [64][Kpad], k = tap*Cin+ci, zero pad
__global__ __launch_bounds__(256) void cvt_weight_kernel(
    const float* __restrict__ w, __bf16* __restrict__ outp,
    int Cin, int ntaps, int Kdim, int Kpad)
{
  size_t i = (size_t)blockIdx.x * 256 + threadIdx.x;
  size_t tot = (size_t)64 * Kpad;
  if (i >= tot) return;
  int co = (int)(i / Kpad);
  int k  = (int)(i - (size_t)co * Kpad);
  float v = 0.f;
  if (k < Kdim) {
    int tap = k / Cin;
    int ci  = k - tap * Cin;
    v = w[((size_t)co * Cin + ci) * ntaps + tap];
  }
  outp[i] = (__bf16)v;
}

static inline int ceil_div(int a, int b) { return (a + b - 1) / b; }

extern "C" void kernel_launch(void* const* d_in, const int* in_sizes, int n_in,
                              void* d_out, int out_size, void* d_ws, size_t ws_size,
                              hipStream_t stream)
{
  // Input order: x may come first or last (top-level dict flatten order unknown).
  int xIdx, pBase;
  if (in_sizes[0] == SEQ_ * B_ * C0 * S_) { xIdx = 0; pBase = 1; }
  else { pBase = 0; xIdx = n_in - 1; }
  const float* x = (const float*)d_in[xIdx];

  // Per-layer param dict, JAX sorted-key order:
  // 0 b111,1 co_w,2 hg_w,3 hi_w,4 ho_w,5 hr_w,6 ln_b,7 ln_g,8 mfp_w,9 mgp_w,
  // 10 mip_w,11 mo_w,12 w111,13 xfp_b,14 xfp_w,15 xg_b,16 xg_w,17 xgp_b,18 xgp_w,
  // 19 xi_b,20 xi_w,21 xip_b,22 xip_w,23 xo_b,24 xo_w,25 xr_b,26 xr_w
  const int XW[7]  = {26,20,16,22,18,14,24};   // xr,xi,xg,xip,xgp,xfp,xo
  const int XB[7]  = {25,19,15,21,17,13,23};
  const int HW9[9] = {5,3,2,10,9,8,4,1,11};    // hr,hi,hg,mip,mgp,mfp,ho,co,mo

  const float *xw[NL_][7], *xb[NL_][7], *hw[NL_][9];
  const float *lng[NL_], *lnb[NL_], *w111[NL_], *b111[NL_];
  for (int l = 0; l < NL_; ++l) {
    int base = pBase + l * 27;
    for (int j = 0; j < 7; ++j) {
      xw[l][j] = (const float*)d_in[base + XW[j]];
      xb[l][j] = (const float*)d_in[base + XB[j]];
    }
    for (int j = 0; j < 9; ++j) hw[l][j] = (const float*)d_in[base + HW9[j]];
    b111[l] = (const float*)d_in[base + 0];
    lnb[l]  = (const float*)d_in[base + 6];
    lng[l]  = (const float*)d_in[base + 7];
    w111[l] = (const float*)d_in[base + 12];
  }

  // -------- workspace carve-up --------
  char* wp = (char*)d_ws;
  auto alloc = [&](size_t bytes) -> char* {
    char* p = wp; wp += (bytes + 255) & ~(size_t)255; return p;
  };
  float* hist  = (float*)alloc((size_t)NL_ * TAU_ * BCS_ * 4);
  float* hbuf0 = (float*)alloc((size_t)BCS_ * 4);
  float* hbuf1 = (float*)alloc((size_t)BCS_ * 4);
  float* mbuf  = (float*)alloc((size_t)BCS_ * 4);
  float* mzero = (float*)alloc((size_t)BCS_ * 4);
  float* pre[7]; for (int i = 0; i < 7; ++i) pre[i] = (float*)alloc((size_t)B_*64*PPI_*4);
  float* gt[7];  for (int i = 0; i < 7; ++i) gt[i]  = (float*)alloc((size_t)BCS_*4);
  float* rec   = (float*)alloc((size_t)BCS_ * 4);
  float* ln4o  = (float*)alloc((size_t)BCS_ * 4);
  float* c111  = (float*)alloc((size_t)BCS_ * 4);
  float* scores = (float*)alloc(64);
  float* attn   = (float*)alloc(64);
  __bf16* xnd  = (__bf16*)alloc((size_t)B_*S_*64*2);
  __bf16* hnd  = (__bf16*)alloc((size_t)B_*S_*64*2);
  __bf16* mnd  = (__bf16*)alloc((size_t)B_*S_*64*2);
  __bf16* cnd  = (__bf16*)alloc((size_t)B_*S_*64*2);
  __bf16* cmnd = (__bf16*)alloc((size_t)B_*S_*128*2);
  const int KPX[NL_]  = {448, 1728};   // Kpad for x-convs (Cin*27 padded to %32)
  const int KDX[NL_]  = {432, 1728};
  const int CINX[NL_] = {16, 64};
  __bf16* xwb[NL_][7]; __bf16* hwb[NL_][9]; __bf16* w111b[NL_];
  for (int l = 0; l < NL_; ++l) {
    for (int j = 0; j < 7; ++j) xwb[l][j] = (__bf16*)alloc((size_t)64 * KPX[l] * 2);
    for (int j = 0; j < 9; ++j) hwb[l][j] = (__bf16*)alloc((size_t)64 * 1728 * 2);
    w111b[l] = (__bf16*)alloc((size_t)64 * 128 * 2);
  }
  (void)ws_size; (void)out_size;

  // -------- weight preprocessing (bf16, k-reordered) --------
  for (int l = 0; l < NL_; ++l) {
    for (int j = 0; j < 7; ++j)
      cvt_weight_kernel<<<ceil_div(64 * KPX[l], 256), 256, 0, stream>>>(
          xw[l][j], xwb[l][j], CINX[l], 27, KDX[l], KPX[l]);
    for (int j = 0; j < 9; ++j)
      cvt_weight_kernel<<<ceil_div(64 * 1728, 256), 256, 0, stream>>>(
          hw[l][j], hwb[l][j], 64, 27, 1728, 1728);
    cvt_weight_kernel<<<ceil_div(64 * 128, 256), 256, 0, stream>>>(
        w111[l], w111b[l], 128, 1, 128, 128);
  }

  // -------- zero state every call (deterministic across graph replays) --------
  fill0_kernel<<<(int)(((size_t)NL_*TAU_*BCS_ + 255) / 256), 256, 0, stream>>>(
      hist, (size_t)NL_ * TAU_ * BCS_);
  fill0_kernel<<<ceil_div(BCS_, 256), 256, 0, stream>>>(hbuf0, (size_t)BCS_);
  fill0_kernel<<<ceil_div(BCS_, 256), 256, 0, stream>>>(hbuf1, (size_t)BCS_);
  fill0_kernel<<<ceil_div(BCS_, 256), 256, 0, stream>>>(mzero, (size_t)BCS_);

  const int gridConv = ceil_div(NPOS_, 128);     // 57
  const int grid111  = ceil_div(B_ * S_, 128);   // 128
  const int gridBCS  = ceil_div(BCS_, 256);
  const int acts6[6] = {0, 0, 1, 0, 1, 0};       // r,i sig; g tanh; ip sig; gp tanh; fp sig

  // x-conv launcher differs per layer (Cin 16 vs 64)
  auto launch_xconv = [&](int l, const __bf16* A, const __bf16* X,
                          const float* bias, float* out, int acc) {
    if (l == 0)
      CONV_X16<<<gridConv, 256, 0, stream>>>(A, X, bias, out, acc);
    else
      CONV_H64<<<gridConv, 256, 0, stream>>>(A, X, bias, out, acc);
  };

  for (int t = 0; t < SEQ_; ++t) {
    for (int l = 0; l < NL_; ++l) {
      const float* xin = (l == 0) ? (x + (size_t)t * B_ * C0 * S_) : hbuf0;
      const int cinx = CINX[l];
      float* hcur = (l == 0) ? hbuf0 : hbuf1;
      const float* min_ = (t == 0) ? mzero : mbuf;   // torch re-inits m=0 per cell at t==0
      float* histL = hist + (size_t)l * TAU_ * BCS_;
      float* cptr  = histL + (size_t)(t % TAU_) * BCS_;               // oldest slot <- c_t
      float* clast = histL + (size_t)((t + TAU_ - 1) % TAU_) * BCS_;  // c_{t-1}

      // activations -> bf16 NDHWC snapshots
      cvt_ndhwc_kernel<<<ceil_div(B_ * cinx * S_, 256), 256, 0, stream>>>(xin, xnd, cinx, cinx, 0);
      cvt_ndhwc_kernel<<<gridBCS, 256, 0, stream>>>(hcur, hnd, 64, 64, 0);
      cvt_ndhwc_kernel<<<gridBCS, 256, 0, stream>>>(min_, mnd, 64, 64, 0);

      // r,i,g: conv(x,W) + conv(h,W)
      for (int g3 = 0; g3 < 3; ++g3) {
        launch_xconv(l, xwb[l][g3], xnd, xb[l][g3], pre[g3], 0);
        CONV_H64<<<gridConv, 256, 0, stream>>>(hwb[l][g3], hnd, nullptr, pre[g3], 1);
      }
      // ip,gp,fp: conv(x,W) + conv(m,W)
      for (int g3 = 0; g3 < 3; ++g3) {
        launch_xconv(l, xwb[l][3+g3], xnd, xb[l][3+g3], pre[3+g3], 0);
        CONV_H64<<<gridConv, 256, 0, stream>>>(hwb[l][3+g3], mnd, nullptr, pre[3+g3], 1);
      }
      for (int gi = 0; gi < 6; ++gi)
        gate_ln_act_kernel<<<B_ * HIDN, 256, 0, stream>>>(pre[gi], gt[gi], acts6[gi]);

      // attention over history (slot order irrelevant: softmax is permutation-invariant)
      attn_scores_kernel<<<B_ * TAU_, 256, 0, stream>>>(gt[0], histL, scores);
      softmax_tau_kernel<<<B_, 32, 0, stream>>>(scores, attn);
      recall_kernel<<<gridBCS, 256, 0, stream>>>(histL, attn, rec);

      // c = i*g + LN4(c_{t-1} + recall); write c_t into the oldest hist slot
      ln4_kernel<<<B_, 256, 0, stream>>>(clast, rec, lng[l], lnb[l], ln4o);
      fma3_kernel<<<gridBCS, 256, 0, stream>>>(gt[1], gt[2], ln4o, cptr, BCS_);

      // m = ip*gp + fp*m_in
      m_update_kernel<<<gridBCS, 256, 0, stream>>>(gt[3], gt[4], gt[5], min_, mbuf, BCS_);

      // new c,m -> bf16 for o-gate convs
      cvt_ndhwc_kernel<<<gridBCS, 256, 0, stream>>>(cptr, cnd, 64, 64, 0);
      cvt_ndhwc_kernel<<<gridBCS, 256, 0, stream>>>(mbuf, mnd, 64, 64, 0);

      // o = sigmoid(LN3(conv(x)+conv(h)+conv(c)+conv(m)))
      launch_xconv(l, xwb[l][6], xnd, xb[l][6], pre[6], 0);
      CONV_H64<<<gridConv, 256, 0, stream>>>(hwb[l][6], hnd, nullptr, pre[6], 1);  // ho
      CONV_H64<<<gridConv, 256, 0, stream>>>(hwb[l][7], cnd, nullptr, pre[6], 1);  // co
      CONV_H64<<<gridConv, 256, 0, stream>>>(hwb[l][8], mnd, nullptr, pre[6], 1);  // mo
      gate_ln_act_kernel<<<B_ * HIDN, 256, 0, stream>>>(pre[6], gt[6], 0);

      // h = o * tanh(conv1x1x1(concat(c,m)))
      cvt_ndhwc_kernel<<<gridBCS, 256, 0, stream>>>(cptr, cmnd, 64, 128, 0);
      cvt_ndhwc_kernel<<<gridBCS, 256, 0, stream>>>(mbuf, cmnd, 64, 128, 64);
      CONV_1X1<<<grid111, 256, 0, stream>>>(w111b[l], cmnd, b111[l], c111, 0);
      h_update_kernel<<<gridBCS, 256, 0, stream>>>(gt[6], c111, hcur, BCS_);
    }
  }

  hipMemcpyAsync(d_out, hbuf1, (size_t)BCS_ * sizeof(float),
                 hipMemcpyDeviceToDevice, stream);
}
